// Model_84353157693506
// MI455X (gfx1250) — matrix operand order repaired and checked by reading, not compiled, not run on hardware
//
#include <hip/hip_runtime.h>
#include <hip/hip_bf16.h>

// ---------------------------------------------------------------------------
// HGP-SL GCN forward for MI455X (gfx1250, wave32).
// Heavy batched matmuls -> v_wmma_f32_16x16x32_bf16 (f32 accumulate) with
// LDS-staged B panels (bf16, transposed, bank-conflict-free pitch) so the
// large adjacency operand streams from HBM exactly once (roofline-bound).
// B fragments are double-buffered in registers so ds_load latency overlaps
// WMMA issue. Everything else -> memory-bound wave32 kernels.
// ---------------------------------------------------------------------------

typedef __attribute__((ext_vector_type(16))) __bf16 v16bf;
typedef __attribute__((ext_vector_type(8)))  float  v8f;

#define WAVES_PER_BLOCK 8

__device__ __forceinline__ float wave_sum(float v) {
#pragma unroll
  for (int o = 16; o >= 1; o >>= 1) v += __shfl_xor(v, o, 32);
  return v;
}
__device__ __forceinline__ float wave_max(float v) {
#pragma unroll
  for (int o = 16; o >= 1; o >>= 1) v = fmaxf(v, __shfl_xor(v, o, 32));
  return v;
}

// ---------------------------------------------------------------------------
// Batched GEMM: C[b] (MxN) = A[b] (MxK) * B[b] (KxN), f32 in/out, bf16 WMMA.
// Requires: N == 256, M % 128 == 0, K % 32 == 0 (true for every call site).
//
// Block = 8 waves = 128 output rows x 256 cols. Per 32-wide K-step:
//   * block stages B slab (32x256) into LDS as bf16, transposed to
//     [n][k] with 20-dword pitch (16B-aligned rows, conflict-free frags)
//   * each wave loads its A fragment (16x32) once, then runs 16 WMMAs
//     against the shared slab; B fragments are double-buffered so the
//     ds_load for tile nj+1 is in flight while tile nj's WMMA executes.
//
// VGPR layouts used here (contiguous K for A and B, not the ISA table order):
//   A 16x32 bf16: lane m=l&15, half=l>>4; element e: K=16*half+e
//   B 32x16 bf16: lane n=l&15, half=l>>4; element e: K=16*half+e
//   C 16x16 f32:  vgpr r: row = r + 8*half, col = l&15
// ---------------------------------------------------------------------------
#define BPITCH 20  // dwords per LDS row (16 data + 4 pad): n*20 % 64 distinct

__device__ __forceinline__ void load_bfrag(v16bf* dst, const unsigned* ldsB,
                                           int base) {
#pragma unroll
  for (int d = 0; d < 8; ++d) ((unsigned*)dst)[d] = ldsB[base + d];
}

__global__ __launch_bounds__(256)
void gemm_bf16_wmma(const float* __restrict__ Ab, long sA, int lda,
                    const float* __restrict__ Bb, long sB, int ldb,
                    float* __restrict__ Cb, long sC, int ldc,
                    int M, int N, int K) {
  (void)M; (void)N;
  __shared__ unsigned ldsB[256 * BPITCH];          // 20 KB

  const int t    = threadIdx.x;                    // 0..255, owns column n=t
  const int wave = t >> 5;
  const int lane = t & 31;
  const int m    = lane & 15;
  const int half = lane >> 4;
  const int b    = blockIdx.y;
  const int m0   = blockIdx.x * 128;

  const float* Arow = Ab + (long)b * sA + (long)(m0 + wave * 16 + m) * lda;
  const float* Bcol = Bb + (long)b * sB + t;

  v8f acc[16];
#pragma unroll
  for (int nj = 0; nj < 16; ++nj) acc[nj] = (v8f){};

  for (int k0 = 0; k0 < K; k0 += 32) {
    // ---- stage B slab: thread t converts column n=t to bf16, K-contiguous --
    const float* bp = Bcol + (long)k0 * ldb;
    __builtin_prefetch(bp + (long)32 * ldb, 0, 1);          // next B slab
    unsigned rowpk[16];
#pragma unroll
    for (int i = 0; i < 16; ++i) {
      const float e0 = bp[(long)(2 * i) * ldb];
      const float e1 = bp[(long)(2 * i + 1) * ldb];
      const __bf16 c0 = (__bf16)e0, c1 = (__bf16)e1;
      rowpk[i] = (unsigned)__builtin_bit_cast(unsigned short, c0) |
                 ((unsigned)__builtin_bit_cast(unsigned short, c1) << 16);
    }
#pragma unroll
    for (int i = 0; i < 4; ++i)
      *(uint4*)&ldsB[t * BPITCH + 4 * i] =
          make_uint4(rowpk[4 * i], rowpk[4 * i + 1], rowpk[4 * i + 2],
                     rowpk[4 * i + 3]);
    __syncthreads();

    // ---- A fragment: 8x float2, convert to bf16 ---------------------------
    const float* ap = Arow + k0 + 16 * half;
    __builtin_prefetch(Arow + k0 + 32, 0, 1);               // next A slab
    v16bf av;
#pragma unroll
    for (int vg = 0; vg < 8; ++vg) {
      const int kk = 2 * vg;
      float2 pr = *(const float2*)(ap + kk);
      av[2 * vg]     = (__bf16)pr.x;
      av[2 * vg + 1] = (__bf16)pr.y;
    }

    // ---- 16 WMMAs, B fragments double-buffered ----------------------------
    const int fbase = m * BPITCH + half * 8;                // 16B aligned
    v16bf bv[2];
    load_bfrag(&bv[0], ldsB, fbase);                        // tile 0
#pragma unroll
    for (int nj = 0; nj < 16; ++nj) {
      const int cur = nj & 1;
      if (nj < 15)                                          // prefetch nj+1
        load_bfrag(&bv[cur ^ 1], ldsB, fbase + (nj + 1) * 16 * BPITCH);
      acc[nj] = __builtin_amdgcn_wmma_f32_16x16x32_bf16(false, av, false,
                                                        bv[cur], (short)0,
                                                        acc[nj], false, false);
    }
    __syncthreads();
  }

  float* cp = Cb + (long)b * sC + (long)(m0 + wave * 16) * ldc + m;
#pragma unroll
  for (int nj = 0; nj < 16; ++nj)
#pragma unroll
    for (int r = 0; r < 8; ++r)
      cp[(long)(r + 8 * half) * ldc + nj * 16] = acc[nj][r];
}

// ---------------------------------------------------------------------------
// Row sums of an [rows, n] matrix -> dinv. mode 0: rsqrt(sum+1) (GCN, A+I),
// mode 1: sum>0 ? 1/sum : 0 (info score). One wave per row.
// ---------------------------------------------------------------------------
__global__ __launch_bounds__(256)
void rowsum_dinv(const float* __restrict__ A, int n, float* __restrict__ dinv,
                 int mode) {
  const int row  = blockIdx.x * WAVES_PER_BLOCK + (threadIdx.x >> 5);
  const int lane = threadIdx.x & 31;
  const float* p = A + (long)row * n;
  float s = 0.f;
  for (int i = lane; i < n; i += 32) s += p[i];
  s = wave_sum(s);
  if (lane == 0)
    dinv[row] = (mode == 0) ? rsqrtf(s + 1.0f) : (s > 0.f ? 1.0f / s : 0.0f);
}

// Y[row, f] *= dinv[row]
__global__ __launch_bounds__(256)
void scale_rows(float* __restrict__ Y, const float* __restrict__ dinv, int F,
                long total) {
  long i = (long)blockIdx.x * blockDim.x + threadIdx.x;
  if (i >= total) return;
  Y[i] *= dinv[i / F];
}

// Z = relu(dinv[row] * (Z + Y) + bias[f])   (adds the A+I self-loop term Y)
__global__ __launch_bounds__(256)
void gcn_epilogue(float* __restrict__ Z, const float* __restrict__ Y,
                  const float* __restrict__ dinv, const float* __restrict__ bias,
                  int F, long total) {
  long i = (long)blockIdx.x * blockDim.x + threadIdx.x;
  if (i >= total) return;
  const int f = (int)(i % F);
  float v = dinv[i / F] * (Z[i] + Y[i]) + bias[f];
  Z[i] = fmaxf(v, 0.f);
}

// score[row] = sum_f | H[row,f] - dinv[row]*T[row,f] |  (T = adj @ H)
__global__ __launch_bounds__(256)
void info_score_kernel(const float* __restrict__ H, const float* __restrict__ T,
                       const float* __restrict__ dinv, float* __restrict__ score,
                       int F) {
  const int row  = blockIdx.x * WAVES_PER_BLOCK + (threadIdx.x >> 5);
  const int lane = threadIdx.x & 31;
  const float d  = dinv[row];
  const float* h = H + (long)row * F;
  const float* t = T + (long)row * F;
  float s = 0.f;
  for (int f = lane; f < F; f += 32) s += fabsf(h[f] - d * t[f]);
  s = wave_sum(s);
  if (lane == 0) score[row] = s;
}

// Per-batch descending bitonic sort of (score, index) in LDS; emit top-k ids.
// (Only the selected *set* matters downstream: readout is max/mean and all
//  later ops are permutation-consistent.)
__global__ void topk_sort(const float* __restrict__ score, int n, int k,
                          int* __restrict__ idx_out) {
  __shared__ float s[1024];
  __shared__ int   id[1024];
  const int b = blockIdx.x, t = threadIdx.x;
  s[t] = score[(long)b * n + t];
  id[t] = t;
  for (int size = 2; size <= n; size <<= 1) {
    for (int stride = size >> 1; stride > 0; stride >>= 1) {
      __syncthreads();
      const int j = t ^ stride;
      if (j > t) {
        const bool descHere = ((t & size) == 0);
        const float a = s[t], c = s[j];
        if (descHere ? (a < c) : (a > c)) {
          s[t] = c; s[j] = a;
          const int tmp = id[t]; id[t] = id[j]; id[j] = tmp;
        }
      }
    }
  }
  __syncthreads();
  if (t < k) idx_out[(long)b * k + t] = id[t];
}

// Xk[b,p,f] = H[b, idx[b,p], f]
__global__ __launch_bounds__(256)
void gather_rows(const float* __restrict__ H, const int* __restrict__ idx,
                 float* __restrict__ Xk, int n_src, int k, int F, long total) {
  long i = (long)blockIdx.x * blockDim.x + threadIdx.x;
  if (i >= total) return;
  const int f = (int)(i % F);
  long t = i / F;
  const int p = (int)(t % k);
  const int b = (int)(t / k);
  Xk[i] = H[((long)b * n_src + idx[(long)b * k + p]) * F + f];
}

// lvec[row] = Xk[row,:] . att[0:F], rvec[row] = Xk[row,:] . att[F:2F]
__global__ __launch_bounds__(256)
void att_proj(const float* __restrict__ Xk, const float* __restrict__ att,
              float* __restrict__ lv, float* __restrict__ rv, int F) {
  const int row  = blockIdx.x * WAVES_PER_BLOCK + (threadIdx.x >> 5);
  const int lane = threadIdx.x & 31;
  const float* p = Xk + (long)row * F;
  float l = 0.f, r = 0.f;
  for (int f = lane; f < F; f += 32) {
    const float v = p[f];
    l += v * att[f];
    r += v * att[F + f];
  }
  l = wave_sum(l); r = wave_sum(r);
  if (lane == 0) { lv[row] = l; rv[row] = r; }
}

// logits[b,p,q] = lv[b,p] + rv[b,q] + LAMB * adj[b, idx[p], idx[q]]
__global__ __launch_bounds__(256)
void build_logits(const float* __restrict__ adj, long sAb, int n_src,
                  const int* __restrict__ idx, const float* __restrict__ lv,
                  const float* __restrict__ rv, float* __restrict__ out, int k,
                  long total) {
  long i = (long)blockIdx.x * blockDim.x + threadIdx.x;
  if (i >= total) return;
  const int q = (int)(i % k);
  long t = i / k;
  const int p = (int)(t % k);
  const int b = (int)(t / k);
  const int ip = idx[(long)b * k + p];
  const int iq = idx[(long)b * k + q];
  const float a = adj[(long)b * sAb + (long)ip * n_src + iq];
  out[i] = lv[(long)b * k + p] + rv[(long)b * k + q] + 1.0f /*LAMB*/ * a;
}

// Row softmax, wave per row, row cached in registers (k <= 512).
__global__ __launch_bounds__(256)
void softmax_rows(float* __restrict__ A, int k) {
  const int row  = blockIdx.x * WAVES_PER_BLOCK + (threadIdx.x >> 5);
  const int lane = threadIdx.x & 31;
  float* p = A + (long)row * k;
  const int cnt = k >> 5;       // 8 or 16
  float v[16];
  float mx = -3.402823466e38f;
  for (int i = 0; i < cnt; ++i) { v[i] = p[lane + 32 * i]; mx = fmaxf(mx, v[i]); }
  mx = wave_max(mx);
  float sum = 0.f;
  for (int i = 0; i < cnt; ++i) { v[i] = __expf(v[i] - mx); sum += v[i]; }
  sum = wave_sum(sum);
  const float inv = 1.0f / sum;
  for (int i = 0; i < cnt; ++i) p[lane + 32 * i] = v[i] * inv;
}

__global__ __launch_bounds__(256)
void zero_f(float* __restrict__ p, long n) {
  long i = (long)blockIdx.x * blockDim.x + threadIdx.x;
  if (i < n) p[i] = 0.f;
}

// sacc[b, f]     += relu(max_p Xk[b,p,f])
// sacc[b, F + f] += relu(mean_p Xk[b,p,f])
__global__ __launch_bounds__(256)
void readout_acc(const float* __restrict__ Xk, float* __restrict__ sacc, int k,
                 int F, int total) {
  const int i = blockIdx.x * blockDim.x + threadIdx.x;
  if (i >= total) return;
  const int b = i / F, f = i % F;
  const float* p = Xk + (long)b * k * F + f;
  float mx = -3.402823466e38f, sum = 0.f;
  for (int r = 0; r < k; ++r) {
    const float v = p[(long)r * F];
    mx = fmaxf(mx, v);
    sum += v;
  }
  sacc[(long)b * 2 * F + f]     += fmaxf(mx, 0.f);
  sacc[(long)b * 2 * F + F + f] += fmaxf(sum / (float)k, 0.f);
}

// out[b,c] = sacc[b,:] . Wlin[:,c] + blin[c]
__global__ __launch_bounds__(256)
void final_linear(const float* __restrict__ s, const float* __restrict__ W,
                  const float* __restrict__ bias, float* __restrict__ out,
                  int Fin, int C, int total) {
  const int i = blockIdx.x * blockDim.x + threadIdx.x;
  if (i >= total) return;
  const int b = i / C, c = i % C;
  float acc = bias[c];
  for (int f = 0; f < Fin; ++f) acc += s[(long)b * Fin + f] * W[(long)f * C + c];
  out[i] = acc;
}

// ---------------------------------------------------------------------------
extern "C" void kernel_launch(void* const* d_in, const int* in_sizes, int n_in,
                              void* d_out, int out_size, void* d_ws, size_t ws_size,
                              hipStream_t stream) {
  (void)in_sizes; (void)n_in; (void)out_size; (void)ws_size;
  const int B = 32, N1 = 1024, FIN = 128, F = 256, K1 = 512, K2 = 256, NCLS = 10;

  const float* x    = (const float*)d_in[0];
  const float* adj  = (const float*)d_in[1];
  const float* W1   = (const float*)d_in[2];
  const float* b1   = (const float*)d_in[3];
  const float* W2   = (const float*)d_in[4];
  const float* b2   = (const float*)d_in[5];
  const float* W3   = (const float*)d_in[6];
  const float* b3   = (const float*)d_in[7];
  const float* att1 = (const float*)d_in[8];
  const float* att2 = (const float*)d_in[9];
  const float* Wlin = (const float*)d_in[10];
  const float* blin = (const float*)d_in[11];
  float* out = (float*)d_out;

  const long big = (long)B * N1 * F;       // 8,388,608 floats (32 MB)
  float* ws    = (float*)d_ws;
  float* bufA  = ws;
  float* bufB  = bufA + big;
  float* bufC  = bufB + big;               // also holds a1 [B,512,512] (same size)
  float* dinvG = bufC + big;
  float* dinvI = dinvG + (long)B * N1;
  float* score = dinvI + (long)B * N1;
  int*   idx   = (int*)(score + (long)B * N1);
  float* lvec  = (float*)(idx + (long)B * K1);
  float* rvec  = lvec + (long)B * K1;
  float* sacc  = rvec + (long)B * K1;      // [B, 2F] = [32, 512]

  auto gemm = [&](const float* A, long sA, int lda, const float* Bm, long sB,
                  int ldb, float* C, long sC, int ldc, int M, int N, int K) {
    dim3 grid(M / 128, B);                 // N==256 always; M%128==0 always
    gemm_bf16_wmma<<<grid, 256, 0, stream>>>(A, sA, lda, Bm, sB, ldb, C, sC, ldc,
                                             M, N, K);
  };
  auto ew = [&](long total) { return dim3((unsigned)((total + 255) / 256)); };

  zero_f<<<ew((long)B * 2 * F), 256, 0, stream>>>(sacc, (long)B * 2 * F);

  // ---------------- Stage 1: GCN1 on (x, adj), pool to K1=512 ----------------
  rowsum_dinv<<<(B * N1) / WAVES_PER_BLOCK, 256, 0, stream>>>(adj, N1, dinvG, 0);
  // XW1 = x @ W1  -> bufA   [B,1024,256]
  gemm(x, (long)N1 * FIN, FIN, W1, 0, F, bufA, (long)N1 * F, F, N1, F, FIN);
  scale_rows<<<ew((long)B * N1 * F), 256, 0, stream>>>(bufA, dinvG, F, (long)B * N1 * F);
  // Z = adj @ Y -> bufB
  gemm(adj, (long)N1 * N1, N1, bufA, (long)N1 * F, F, bufB, (long)N1 * F, F, N1, F, N1);
  gcn_epilogue<<<ew((long)B * N1 * F), 256, 0, stream>>>(bufB, bufA, dinvG, b1, F,
                                                         (long)B * N1 * F);   // h1 in bufB
  rowsum_dinv<<<(B * N1) / WAVES_PER_BLOCK, 256, 0, stream>>>(adj, N1, dinvI, 1);
  // T = adj @ h1 -> bufA
  gemm(adj, (long)N1 * N1, N1, bufB, (long)N1 * F, F, bufA, (long)N1 * F, F, N1, F, N1);
  info_score_kernel<<<(B * N1) / WAVES_PER_BLOCK, 256, 0, stream>>>(bufB, bufA, dinvI,
                                                                    score, F);
  topk_sort<<<B, N1, 0, stream>>>(score, N1, K1, idx);
  // xk -> bufA [B,512,256]
  gather_rows<<<ew((long)B * K1 * F), 256, 0, stream>>>(bufB, idx, bufA, N1, K1, F,
                                                        (long)B * K1 * F);
  att_proj<<<(B * K1) / WAVES_PER_BLOCK, 256, 0, stream>>>(bufA, att1, lvec, rvec, F);
  build_logits<<<ew((long)B * K1 * K1), 256, 0, stream>>>(adj, (long)N1 * N1, N1, idx,
                                                          lvec, rvec, bufC, K1,
                                                          (long)B * K1 * K1);
  softmax_rows<<<(B * K1) / WAVES_PER_BLOCK, 256, 0, stream>>>(bufC, K1);  // a1 in bufC
  readout_acc<<<ew(B * F), 256, 0, stream>>>(bufA, sacc, K1, F, B * F);

  // ---------------- Stage 2: GCN2 on (xk=bufA, a1=bufC), pool to K2 ----------
  rowsum_dinv<<<(B * K1) / WAVES_PER_BLOCK, 256, 0, stream>>>(bufC, K1, dinvG, 0);
  rowsum_dinv<<<(B * K1) / WAVES_PER_BLOCK, 256, 0, stream>>>(bufC, K1, dinvI, 1);
  // XW2 = xk @ W2 -> bufB
  gemm(bufA, (long)K1 * F, F, W2, 0, F, bufB, (long)K1 * F, F, K1, F, F);
  scale_rows<<<ew((long)B * K1 * F), 256, 0, stream>>>(bufB, dinvG, F, (long)B * K1 * F);
  // Z2 = a1 @ Y2 -> bufA (xk dead)
  gemm(bufC, (long)K1 * K1, K1, bufB, (long)K1 * F, F, bufA, (long)K1 * F, F, K1, F, K1);
  gcn_epilogue<<<ew((long)B * K1 * F), 256, 0, stream>>>(bufA, bufB, dinvG, b2, F,
                                                         (long)B * K1 * F);   // h2 in bufA
  // T2 = a1 @ h2 -> bufB
  gemm(bufC, (long)K1 * K1, K1, bufA, (long)K1 * F, F, bufB, (long)K1 * F, F, K1, F, K1);
  info_score_kernel<<<(B * K1) / WAVES_PER_BLOCK, 256, 0, stream>>>(bufA, bufB, dinvI,
                                                                    score, F);
  topk_sort<<<B, K1, 0, stream>>>(score, K1, K2, idx);
  // xk2 -> bufB [B,256,256]
  gather_rows<<<ew((long)B * K2 * F), 256, 0, stream>>>(bufA, idx, bufB, K1, K2, F,
                                                        (long)B * K2 * F);
  att_proj<<<(B * K2) / WAVES_PER_BLOCK, 256, 0, stream>>>(bufB, att2, lvec, rvec, F);
  // a2 logits -> bufA (h2 dead)
  build_logits<<<ew((long)B * K2 * K2), 256, 0, stream>>>(bufC, (long)K1 * K1, K1, idx,
                                                          lvec, rvec, bufA, K2,
                                                          (long)B * K2 * K2);
  softmax_rows<<<(B * K2) / WAVES_PER_BLOCK, 256, 0, stream>>>(bufA, K2);  // a2 in bufA
  readout_acc<<<ew(B * F), 256, 0, stream>>>(bufB, sacc, K2, F, B * F);

  // ---------------- Stage 3: GCN3 on (xk2=bufB, a2=bufA), readout only -------
  rowsum_dinv<<<(B * K2) / WAVES_PER_BLOCK, 256, 0, stream>>>(bufA, K2, dinvG, 0);
  // XW3 = xk2 @ W3 -> bufC (a1 dead)
  gemm(bufB, (long)K2 * F, F, W3, 0, F, bufC, (long)K2 * F, F, K2, F, F);
  scale_rows<<<ew((long)B * K2 * F), 256, 0, stream>>>(bufC, dinvG, F, (long)B * K2 * F);
  // Z3 = a2 @ Y3 -> bufB (xk2 dead after readout)
  gemm(bufA, (long)K2 * K2, K2, bufC, (long)K2 * F, F, bufB, (long)K2 * F, F, K2, F, K2);
  gcn_epilogue<<<ew((long)B * K2 * F), 256, 0, stream>>>(bufB, bufC, dinvG, b3, F,
                                                         (long)B * K2 * F);   // h3 in bufB
  readout_acc<<<ew(B * F), 256, 0, stream>>>(bufB, sacc, K2, F, B * F);

  final_linear<<<ew(B * NCLS), 256, 0, stream>>>(sacc, Wlin, blin, out, 2 * F, NCLS,
                                                 B * NCLS);
}
